// MultiHeadAttention_60078002536549
// MI455X (gfx1250) — matrix operand order repaired and checked by reading, not compiled, not run on hardware
//
#include <hip/hip_runtime.h>

// B=2, H=16, N=2048, HD=64, D=1024, FF=4096  (hardcoded from reference)

typedef __attribute__((ext_vector_type(16))) __bf16 v16bf;
typedef __attribute__((ext_vector_type(8)))  float  v8f;

union FragAB {          // 16 bf16 operands of one WMMA A/B fragment
    uint4 q[2];
    v16bf v;
};

__device__ __forceinline__ v8f vzero() {
    v8f z = {0.f, 0.f, 0.f, 0.f, 0.f, 0.f, 0.f, 0.f};
    return z;
}

__device__ __forceinline__ v8f wmma_bf16(v16bf a, v16bf b, v8f c) {
    // D(16x16 f32) = A(16x32 bf16) x B(32x16 bf16) + C
    return __builtin_amdgcn_wmma_f32_16x16x32_bf16(false, a, false, b,
                                                   (short)0, c, false, false);
}

__device__ __forceinline__ unsigned short f2bf(float f) {
    union { float f; unsigned u; } c;
    c.f = f;
    unsigned u = c.u;
    return (unsigned short)((u + 0x7FFFu + ((u >> 16) & 1u)) >> 16); // RNE
}

__device__ __forceinline__ unsigned pack2bf(float lo, float hi) {
    return (unsigned)f2bf(lo) | ((unsigned)f2bf(hi) << 16);
}

// ---- CDNA5 async global->LDS copy (ASYNCcnt-tracked, no VGPR round trip) --
// GV mode: per-lane 16B  MEM[vaddr] -> LDS[vdst].  Generic LDS pointers carry
// the LDS byte address in their low 32 bits (ISA 10.2 aperture rule).
__device__ __forceinline__ void async_copy_b128(void* lds_dst, const void* gsrc) {
    unsigned ldsa = (unsigned)(unsigned long long)lds_dst;
    unsigned long long ga = (unsigned long long)gsrc;
    asm volatile("global_load_async_to_lds_b128 %0, %1, off"
                 :: "v"(ldsa), "v"(ga) : "memory");
}
__device__ __forceinline__ void wait_async0() {
    asm volatile("s_wait_asynccnt 0x0" ::: "memory");
}

// ---------------------------------------------------------------------------
// Kernel 1: f32 -> bf16 transpose-convert for weights (Wt[n][k] = W[k][n])
// ---------------------------------------------------------------------------
__global__ __launch_bounds__(256) void transpose_cvt_kernel(
    const float* __restrict__ src, unsigned short* __restrict__ dst,
    int K, int Nc)
{
    size_t idx = (size_t)blockIdx.x * 256 + threadIdx.x;
    size_t total = (size_t)K * Nc;
    if (idx >= total) return;
    size_t kk = idx / Nc;
    size_t nn = idx - kk * Nc;
    dst[nn * (size_t)K + kk] = f2bf(src[idx]);   // coalesced read along n
}

// ---------------------------------------------------------------------------
// Kernel 2: K -> bf16 (same layout), V -> bf16 transposed per head:
//   Kb[bh][n][hd] = bf16(K),  Vt[bh][hd][n] = bf16(V)
// ---------------------------------------------------------------------------
__global__ __launch_bounds__(256) void qkv_prep_kernel(
    const float* __restrict__ Kf, const float* __restrict__ Vf,
    unsigned short* __restrict__ Kb, unsigned short* __restrict__ Vt)
{
    size_t idx = (size_t)blockIdx.x * 256 + threadIdx.x;   // < 2*16*2048*64
    Kb[idx] = f2bf(Kf[idx]);
    size_t bh  = idx >> 17;          // / (2048*64)
    size_t rem = idx & 131071;
    size_t n   = rem >> 6;
    size_t hd  = rem & 63;
    Vt[(bh << 17) + (hd << 11) + n] = f2bf(Vf[idx]);
}

// ---------------------------------------------------------------------------
// Kernel 3: flash attention (transposed S/out tiles), fused residual:
//   x1 = x + softmax(QK^T/8)V     written in (B,N,D) layout
// grid (N/64, H, B), block 128 (4 waves, 16 query columns per wave).
// S^T = K(A) x Q^T(B): per-lane query column -> scalar softmax stats;
// K/V tiles double-buffered in LDS via async global->LDS copies.
// ---------------------------------------------------------------------------
__global__ __launch_bounds__(128) void attn_flash_kernel(
    const float* __restrict__ Qg, const unsigned short* __restrict__ Kb,
    const unsigned short* __restrict__ Vt, const float* __restrict__ Xg,
    float* __restrict__ X1)
{
    __shared__ unsigned short Qs[64][72];        // query-major (B operand)
    __shared__ unsigned short Ks[2][32][72];     // key-major   (A operand)
    __shared__ unsigned short Vs[2][64][40];     // channel-major (A operand)
    __shared__ unsigned short Ps[4][16][40];     // P^T per wave: [q][key]

    const int tq = blockIdx.x, head = blockIdx.y, batch = blockIdx.z;
    const int tid  = threadIdx.x;
    const int wid  = tid >> 5;
    const int lane = tid & 31;
    const int l16  = lane & 15;
    const int hi   = lane >> 4;

    const size_t hdbase = ((size_t)(batch * 16 + head)) << 17;   // *2048*64
    const float* Qh          = Qg + hdbase + (size_t)tq * 64 * 64;
    const unsigned short* Kh = Kb + hdbase;
    const unsigned short* Vh = Vt + hdbase;     // [ch][n], n-stride 2048

    // per-thread copy slices
    const int kr = tid >> 2, ke = (tid & 3) << 4;   // K: row 0..31, 16 elems
    const int vc = tid >> 1, ve = (tid & 1) << 4;   // V: ch 0..63, 16 elems

    // ---- stage Q tile (64 x 64) as bf16, softmax scale folded in ----
    for (int i = tid; i < 1024; i += 128) {
        int r = i >> 4, c4 = (i & 15) << 2;
        float4 qv = *(const float4*)&Qh[r * 64 + c4];
        ushort4 o;
        o.x = f2bf(qv.x * 0.125f); o.y = f2bf(qv.y * 0.125f);
        o.z = f2bf(qv.z * 0.125f); o.w = f2bf(qv.w * 0.125f);
        *(ushort4*)&Qs[r][c4] = o;
    }

    // ---- prologue: async-stage K/V block 0 ----
    {
        const unsigned short* gk = Kh + (size_t)kr * 64 + ke;
        async_copy_b128(&Ks[0][kr][ke],     gk);
        async_copy_b128(&Ks[0][kr][ke + 8], gk + 8);
        const unsigned short* gv = Vh + ((size_t)vc << 11) + ve;
        async_copy_b128(&Vs[0][vc][ve],     gv);
        async_copy_b128(&Vs[0][vc][ve + 8], gv + 8);
    }
    wait_async0();
    __syncthreads();

    const int abase = hi * 8;    // A-fragment K base per half-wave
    const int bbase = hi * 8;    // B-fragment K base per half-wave

    // ---- Q^T B-fragments (column = this lane's query) ----
    const int qloc = wid * 16 + l16;
    FragAB qbf[2];
#pragma unroll
    for (int ks = 0; ks < 2; ks++) {
        qbf[ks].q[0] = *(const uint4*)&Qs[qloc][ks * 32 + bbase];
        qbf[ks].q[1] = *(const uint4*)&Qs[qloc][ks * 32 + bbase + 16];
    }

    float mst = -1e30f, lst = 0.f;
    v8f acc[4];                       // out^T: rows = channels, col = query
#pragma unroll
    for (int j = 0; j < 4; j++) acc[j] = vzero();

    for (int kb = 0; kb < 2048 / 32; kb++) {
        const int cur = kb & 1;
        const bool more = (kb + 1) < 2048 / 32;

        // ---- issue async copies of next K/V block into alternate buffer ----
        if (more) {
            const int nxt = cur ^ 1;
            const unsigned short* gk = Kh + (size_t)((kb + 1) * 32 + kr) * 64 + ke;
            async_copy_b128(&Ks[nxt][kr][ke],     gk);
            async_copy_b128(&Ks[nxt][kr][ke + 8], gk + 8);
            const unsigned short* gv = Vh + ((size_t)vc << 11) + (kb + 1) * 32 + ve;
            async_copy_b128(&Vs[nxt][vc][ve],     gv);
            async_copy_b128(&Vs[nxt][vc][ve + 8], gv + 8);
        }

        // ---- S^T = K @ Q^T : two 16-key tiles (rows=keys, col=query) ----
        v8f st[2];
#pragma unroll
        for (int kt = 0; kt < 2; kt++) {
            v8f s = vzero();
            const int key = kt * 16 + l16;
#pragma unroll
            for (int ks = 0; ks < 2; ks++) {
                FragAB kf;
                kf.q[0] = *(const uint4*)&Ks[cur][key][ks * 32 + abase];
                kf.q[1] = *(const uint4*)&Ks[cur][key][ks * 32 + abase + 16];
                s = wmma_bf16(kf.v, qbf[ks].v, s);
            }
            st[kt] = s;
        }

        // ---- online softmax: 16 local keys + one xor-16 combine ----
        float t = st[0][0];
#pragma unroll
        for (int r = 1; r < 8; r++) t = fmaxf(t, st[0][r]);
#pragma unroll
        for (int r = 0; r < 8; r++) t = fmaxf(t, st[1][r]);
        t = fmaxf(t, __shfl_xor(t, 16, 32));
        const float mnew = fmaxf(mst, t);
        const float alpha = __expf(mst - mnew);
        float rs = 0.f;
        uint4 pw[2];
#pragma unroll
        for (int kt = 0; kt < 2; kt++) {
            float p[8];
#pragma unroll
            for (int r = 0; r < 8; r++) {
                p[r] = __expf(st[kt][r] - mnew);
                rs += p[r];
            }
            pw[kt].x = pack2bf(p[0], p[1]);
            pw[kt].y = pack2bf(p[2], p[3]);
            pw[kt].z = pack2bf(p[4], p[5]);
            pw[kt].w = pack2bf(p[6], p[7]);
        }
        rs += __shfl_xor(rs, 16, 32);
        lst = lst * alpha + rs;
        mst = mnew;
#pragma unroll
        for (int j = 0; j < 4; j++)
#pragma unroll
            for (int r = 0; r < 8; r++) acc[j][r] *= alpha;

        // P^T to LDS: 8 consecutive keys per lane -> single b128 store each
        *(uint4*)&Ps[wid][l16][hi * 8]      = pw[0];
        *(uint4*)&Ps[wid][l16][16 + hi * 8] = pw[1];

        // ---- acc^T += V^T(A: 16ch x 32key) @ P^T(B: 32key x 16q) ----
        FragAB pb;
        pb.q[0] = *(const uint4*)&Ps[wid][l16][bbase];
        pb.q[1] = *(const uint4*)&Ps[wid][l16][bbase + 16];
#pragma unroll
        for (int j = 0; j < 4; j++) {
            const int ch = j * 16 + l16;
            FragAB vf;
            vf.q[0] = *(const uint4*)&Vs[cur][ch][abase];
            vf.q[1] = *(const uint4*)&Vs[cur][ch][abase + 16];
            acc[j] = wmma_bf16(vf.v, pb.v, acc[j]);
        }

        if (more) wait_async0();     // next buffer ready before barrier
        __syncthreads();
    }

    // ---- epilogue: each lane owns query row (l16), 8 consecutive channels
    const float rinv = 1.f / lst;
    const size_t row_g = (size_t)(batch * 2048 + tq * 64 + wid * 16 + l16);
#pragma unroll
    for (int j = 0; j < 4; j++) {
        const size_t idx = row_g * 1024 + head * 64 + j * 16 + hi * 8;
        float4 x0 = *(const float4*)&Xg[idx];
        float4 x1v = *(const float4*)&Xg[idx + 4];
        float4 o0, o1;
        o0.x = x0.x + acc[j][0] * rinv;  o0.y = x0.y + acc[j][1] * rinv;
        o0.z = x0.z + acc[j][2] * rinv;  o0.w = x0.w + acc[j][3] * rinv;
        o1.x = x1v.x + acc[j][4] * rinv; o1.y = x1v.y + acc[j][5] * rinv;
        o1.z = x1v.z + acc[j][6] * rinv; o1.w = x1v.w + acc[j][7] * rinv;
        *(float4*)&X1[idx] = o0;
        *(float4*)&X1[idx + 4] = o1;
    }
}

// ---------------------------------------------------------------------------
// Kernel 4: LayerNorm over D=1024, output bf16
// ---------------------------------------------------------------------------
__global__ __launch_bounds__(256) void ln_kernel(
    const float* __restrict__ X1, const float* __restrict__ w,
    const float* __restrict__ b, unsigned short* __restrict__ H)
{
    const int row = blockIdx.x;
    const int tid = threadIdx.x;
    const float* xr = X1 + (size_t)row * 1024;
    float4 vv = *(const float4*)&xr[tid * 4];
    float s  = vv.x + vv.y + vv.z + vv.w;
    float ss = vv.x * vv.x + vv.y * vv.y + vv.z * vv.z + vv.w * vv.w;
#pragma unroll
    for (int off = 16; off; off >>= 1) {
        s  += __shfl_xor(s,  off, 32);
        ss += __shfl_xor(ss, off, 32);
    }
    __shared__ float ps[8], pq[8];
    const int wid = tid >> 5;
    if ((tid & 31) == 0) { ps[wid] = s; pq[wid] = ss; }
    __syncthreads();
    s = 0.f; ss = 0.f;
#pragma unroll
    for (int i = 0; i < 8; i++) { s += ps[i]; ss += pq[i]; }
    const float mu   = s * (1.f / 1024.f);
    const float var  = ss * (1.f / 1024.f) - mu * mu;
    const float rstd = rsqrtf(var + 1e-5f);
    const int c = tid * 4;
    ushort4 o;
    o.x = f2bf((vv.x - mu) * rstd * w[c + 0] + b[c + 0]);
    o.y = f2bf((vv.y - mu) * rstd * w[c + 1] + b[c + 1]);
    o.z = f2bf((vv.z - mu) * rstd * w[c + 2] + b[c + 2]);
    o.w = f2bf((vv.w - mu) * rstd * w[c + 3] + b[c + 3]);
    *(ushort4*)&H[(size_t)row * 1024 + c] = o;
}

// ---------------------------------------------------------------------------
// Kernel 5/6: bf16 WMMA GEMM, C = A[MxK] * Bt^T  (Bt stored [Ncol][K])
// MODE 0: +bias, exact GeLU, bf16 out.  MODE 1: +bias +residual, f32 out.
// block 256 (8 waves 2x4), tile 128x128x32, double-buffered LDS filled by
// async global->LDS copies overlapped with WMMA compute.
// ---------------------------------------------------------------------------
template <int MODE>
__global__ __launch_bounds__(256) void gemm_bf16_kernel(
    const unsigned short* __restrict__ A, const unsigned short* __restrict__ Bt,
    const float* __restrict__ bias, const float* __restrict__ resid,
    void* __restrict__ Cout, int M, int Ncol, int Kdim)
{
    __shared__ unsigned short As[2][128][40];   // padded rows, double buffer
    __shared__ unsigned short Bs[2][128][40];

    const int bn = blockIdx.x, bm = blockIdx.y;
    const int tid  = threadIdx.x;
    const int wid  = tid >> 5;
    const int lane = tid & 31;
    const int l16  = lane & 15;
    const int hi   = lane >> 4;
    const int wm = wid >> 2, wn = wid & 3;   // wave tile: 64 rows x 32 cols

    v8f acc[4][2];
#pragma unroll
    for (int i = 0; i < 4; i++)
#pragma unroll
        for (int j = 0; j < 2; j++) acc[i][j] = vzero();

    const int ldr = tid >> 2;           // 0..63 (two rows per thread)
    const int ldc = (tid & 3) << 3;     // 0,8,16,24 (bf16 elems, 16B chunks)
    const size_t aoff = (size_t)(bm * 128) * Kdim;
    const size_t boff = (size_t)(bn * 128) * Kdim;
    const int ab = hi * 8, bb = hi * 8;
    const int nIter = Kdim >> 5;

    // ---- prologue: async-stage K-tile 0 ----
#pragma unroll
    for (int rr = 0; rr < 2; rr++) {
        const int row = ldr + rr * 64;
        async_copy_b128(&As[0][row][ldc], &A[aoff + (size_t)row * Kdim + ldc]);
        async_copy_b128(&Bs[0][row][ldc], &Bt[boff + (size_t)row * Kdim + ldc]);
    }
    wait_async0();
    __syncthreads();

    for (int it = 0; it < nIter; ++it) {
        const int cur = it & 1;
        const bool more = (it + 1) < nIter;
        const int k1 = (it + 1) << 5;

        if (more) {                      // fill alternate buffer during compute
            const int nxt = cur ^ 1;
#pragma unroll
            for (int rr = 0; rr < 2; rr++) {
                const int row = ldr + rr * 64;
                async_copy_b128(&As[nxt][row][ldc],
                                &A[aoff + (size_t)row * Kdim + k1 + ldc]);
                async_copy_b128(&Bs[nxt][row][ldc],
                                &Bt[boff + (size_t)row * Kdim + k1 + ldc]);
            }
        }

        FragAB af[4], bfr[2];
#pragma unroll
        for (int i = 0; i < 4; i++) {
            const int m = wm * 64 + i * 16 + l16;
            af[i].q[0] = *(const uint4*)&As[cur][m][ab];
            af[i].q[1] = *(const uint4*)&As[cur][m][ab + 16];
        }
#pragma unroll
        for (int j = 0; j < 2; j++) {
            const int n = wn * 32 + j * 16 + l16;
            bfr[j].q[0] = *(const uint4*)&Bs[cur][n][bb];
            bfr[j].q[1] = *(const uint4*)&Bs[cur][n][bb + 16];
        }
#pragma unroll
        for (int i = 0; i < 4; i++)
#pragma unroll
            for (int j = 0; j < 2; j++)
                acc[i][j] = wmma_bf16(af[i].v, bfr[j].v, acc[i][j]);

        if (more) wait_async0();
        __syncthreads();
    }

    // ---- epilogue ----
#pragma unroll
    for (int i = 0; i < 4; i++) {
#pragma unroll
        for (int j = 0; j < 2; j++) {
            const int col = bn * 128 + wn * 32 + j * 16 + l16;
            const float bc = bias[col];
#pragma unroll
            for (int r = 0; r < 8; r++) {
                const int row = bm * 128 + wm * 64 + i * 16 + r + hi * 8;
                const size_t idx = (size_t)row * Ncol + col;
                float val = acc[i][j][r] + bc;
                if (MODE == 0) {
                    float g = 0.5f * val * (1.f + erff(val * 0.70710678118f));
                    ((unsigned short*)Cout)[idx] = f2bf(g);
                } else {
                    ((float*)Cout)[idx] = val + resid[idx];
                }
            }
        }
    }
}

// ---------------------------------------------------------------------------
extern "C" void kernel_launch(void* const* d_in, const int* in_sizes, int n_in,
                              void* d_out, int out_size, void* d_ws, size_t ws_size,
                              hipStream_t stream)
{
    (void)in_sizes; (void)n_in; (void)out_size; (void)ws_size;
    const float* x   = (const float*)d_in[0];
    const float* q   = (const float*)d_in[1];
    const float* k   = (const float*)d_in[2];
    const float* v   = (const float*)d_in[3];
    const float* lnw = (const float*)d_in[4];
    const float* lnb = (const float*)d_in[5];
    const float* w1  = (const float*)d_in[6];
    const float* b1  = (const float*)d_in[7];
    const float* w2  = (const float*)d_in[8];
    const float* b2  = (const float*)d_in[9];
    float* out = (float*)d_out;

    // workspace layout (88 MB total)
    char* ws = (char*)d_ws;
    float*          x1   = (float*)ws;                                  // 16 MB
    unsigned short* hbuf = (unsigned short*)(ws + (16ull << 20));       //  8 MB
    unsigned short* w1t  = (unsigned short*)(ws + (24ull << 20));       //  8 MB
    unsigned short* w2t  = (unsigned short*)(ws + (32ull << 20));       //  8 MB
    unsigned short* G    = (unsigned short*)(ws + (40ull << 20));       // 32 MB
    unsigned short* Kb   = (unsigned short*)(ws + (72ull << 20));       //  8 MB
    unsigned short* Vt   = (unsigned short*)(ws + (80ull << 20));       //  8 MB

    // weights -> bf16 transposed (L2-resident thereafter: 8 MB each)
    transpose_cvt_kernel<<<16384, 256, 0, stream>>>(w1, w1t, 1024, 4096);
    transpose_cvt_kernel<<<16384, 256, 0, stream>>>(w2, w2t, 4096, 1024);

    // K -> bf16, V -> bf16 transposed per head
    qkv_prep_kernel<<<16384, 256, 0, stream>>>(k, v, Kb, Vt);

    // attention + residual -> x1 (B,N,D)
    attn_flash_kernel<<<dim3(32, 16, 2), 128, 0, stream>>>(q, Kb, Vt, x, x1);

    // layernorm -> bf16 h
    ln_kernel<<<4096, 256, 0, stream>>>(x1, lnw, lnb, hbuf);

    // FFN1: G = gelu(h @ W1 + b1), bf16
    gemm_bf16_kernel<0><<<dim3(32, 32), 256, 0, stream>>>(
        hbuf, w1t, b1, nullptr, (void*)G, 4096, 4096, 1024);

    // FFN2: out = G @ W2 + b2 + x1, f32
    gemm_bf16_kernel<1><<<dim3(8, 32), 256, 0, stream>>>(
        G, w2t, b2, x1, (void*)out, 4096, 1024, 4096);
}
